// MultiHeadLayerAttention_2422361555490
// MI455X (gfx1250) — matrix operand order repaired and checked by reading, not compiled, not run on hardware
//
#include <hip/hip_runtime.h>
#include <hip/hip_bf16.h>

typedef __attribute__((ext_vector_type(16))) __bf16 v16bf;
typedef __attribute__((ext_vector_type(8)))  float  v8f;

#define L_DIM    5
#define IN_DIM   256
#define OUT_DIM  256
#define H_DIM    4
#define DH_DIM   64
#define RED_DIM  16
#define SPB      16              // samples per block
#define ROWS     (SPB * L_DIM)   // 80 rows of X per block
#define XPITCH   264             // 256 + 8 bf16 pad (bank-conflict-free A loads)
#define QKVPITCH 776             // 768 + 8 bf16 pad
#define SCALE_F  0.0625f         // 1/sqrt(256)
#define NCT      48              // column tiles (768/16)
#define NKK      8               // k-steps (256/32)
#define WSWZ_ELEMS (NCT * NKK * 32 * 16)   // 196608 bf16 = 393216 B in d_ws

static constexpr int LDS_X_BYTES   = ROWS * XPITCH * 2;    // 42240
static constexpr int LDS_QKV_BYTES = ROWS * QKVPITCH * 2;  // 124160
static constexpr int LDS_TOTAL     = LDS_X_BYTES + LDS_QKV_BYTES; // 166400

union V16 { uint4 u[2]; v16bf v; };

__device__ __forceinline__ __bf16 to_bf16(float f)  { return (__bf16)f; }
__device__ __forceinline__ float  fr_bf16(__bf16 b) { return (float)b; }

// ---------------------------------------------------------------------------
// Prep: swizzle [Wq|Wk|Wv] (fp32, [H,IN,DH]) into WMMA-B lane layout, bf16.
// ws index = (((ct*8 + kk)*32 + lane)*16 + i)
//   lane -> col = ct*16 + (lane&15);  elem i -> K = kk*32 + (lane>>4)*16 + i
// ---------------------------------------------------------------------------
extern "C" __global__ __launch_bounds__(256)
void wswz_kernel(const float* __restrict__ Wq, const float* __restrict__ Wk,
                 const float* __restrict__ Wv, __bf16* __restrict__ wsw)
{
    const int idx = blockIdx.x * 256 + threadIdx.x;
    if (idx >= WSWZ_ELEMS) return;
    const int i    = idx & 15;
    const int lane = (idx >> 4) & 31;
    const int kk   = (idx >> 9) & 7;
    const int ct   = idx >> 12;
    const int col  = ct * 16 + (lane & 15);
    const int K    = kk * 32 + (lane >> 4) * 16 + i;
    const int sel  = col >> 8;            // 0=Q 1=K 2=V
    const int cc   = col & 255;
    const int h    = cc >> 6, d = cc & 63;
    const float* Wsel = (sel == 0) ? Wq : ((sel == 1) ? Wk : Wv);
    wsw[idx] = to_bf16(Wsel[h * IN_DIM * DH_DIM + K * DH_DIM + d]);
}

// ---------------------------------------------------------------------------
// Fused kernel: X tile -> LDS bf16 | WMMA QKV -> LDS | attention | SE | store
// ---------------------------------------------------------------------------
extern "C" __global__ __launch_bounds__(256)
void mhla_fused_kernel(const float* __restrict__ x,
                       const __bf16* __restrict__ wsw,
                       const float* __restrict__ bq, const float* __restrict__ bk,
                       const float* __restrict__ bv,
                       const float* __restrict__ W1, const float* __restrict__ b1,
                       const float* __restrict__ W2, const float* __restrict__ b2,
                       float* __restrict__ out, int N)
{
    extern __shared__ char smem[];
    __bf16* Xs   = (__bf16*)smem;
    __bf16* QKVs = (__bf16*)(smem + LDS_X_BYTES);

    const int tid    = threadIdx.x;
    const int block0 = blockIdx.x * SPB;
    const int vrows  = min(ROWS, (N - block0) * L_DIM);
    const float* xblk = x + (size_t)block0 * L_DIM * IN_DIM;

    // ---------------- Phase 0: X tile (80x256 f32, contiguous) -> bf16 LDS ----
    for (int i = tid; i < ROWS * (IN_DIM / 4); i += 256) {
        const int row = i >> 6;
        const int k4  = (i & 63) << 2;
        float4 v = make_float4(0.f, 0.f, 0.f, 0.f);
        if (row < vrows) v = ((const float4*)xblk)[i];
        __bf16* dst = Xs + row * XPITCH + k4;
        dst[0] = to_bf16(v.x); dst[1] = to_bf16(v.y);
        dst[2] = to_bf16(v.z); dst[3] = to_bf16(v.w);
    }
    __syncthreads();

    // ---------------- Phase 1: QKV[80x768] = X @ Wcat + bias (WMMA bf16) ------
    {
        const int lane = tid & 31;
        const int wave = tid >> 5;
        const int lr   = lane & 15;
        const int lh   = lane >> 4;

        for (int ct = wave; ct < NCT; ct += 8) {
            const int col = ct * 16 + lr;
            const int sel = col >> 8;
            const int cc  = col & 255;
            const int h   = cc >> 6, d = cc & 63;
            const float* bsel = (sel == 0) ? bq : ((sel == 1) ? bk : bv);
            const float  bias = bsel[h * DH_DIM + d];

            // B panel: 8 ksteps x 32B contiguous per lane (pre-swizzled bf16)
            v16bf Bp[NKK];
            #pragma unroll
            for (int kk = 0; kk < NKK; ++kk) {
                const uint4* wp = (const uint4*)(wsw + (((ct * NKK + kk) * 32 + lane) << 4));
                V16 b; b.u[0] = wp[0]; b.u[1] = wp[1];
                Bp[kk] = b.v;
            }

            // M-tile pairs: two independent accumulator chains interleaved so
            // consecutive WMMAs are independent (fills the D->A/B hazard slots).
            #pragma unroll
            for (int mp = 0; mp < 2; ++mp) {
                const int mt0 = mp * 2, mt1 = mp * 2 + 1;
                v8f acc0, acc1;
                #pragma unroll
                for (int r = 0; r < 8; ++r) { acc0[r] = bias; acc1[r] = bias; }
                const int ar0 = mt0 * 16 + lr;
                const int ar1 = mt1 * 16 + lr;
                #pragma unroll
                for (int kk = 0; kk < NKK; ++kk) {
                    const int kb = kk * 32 + lh * 16;
                    const __bf16* ap0 = Xs + ar0 * XPITCH + kb;
                    const __bf16* ap1 = Xs + ar1 * XPITCH + kb;
                    V16 a0, a1;
                    a0.u[0] = *(const uint4*)ap0;
                    a0.u[1] = *(const uint4*)(ap0 + 8);
                    a1.u[0] = *(const uint4*)ap1;
                    a1.u[1] = *(const uint4*)(ap1 + 8);
                    acc0 = __builtin_amdgcn_wmma_f32_16x16x32_bf16(
                            false, a0.v, false, Bp[kk], (short)0, acc0, false, false);
                    acc1 = __builtin_amdgcn_wmma_f32_16x16x32_bf16(
                            false, a1.v, false, Bp[kk], (short)0, acc1, false, false);
                }
                #pragma unroll
                for (int r = 0; r < 8; ++r) {
                    const int row0 = mt0 * 16 + lh * 8 + r;
                    const int row1 = mt1 * 16 + lh * 8 + r;
                    QKVs[row0 * QKVPITCH + col] = to_bf16(acc0[r]);
                    QKVs[row1 * QKVPITCH + col] = to_bf16(acc1[r]);
                }
            }
            // tail M-tile (mt = 4)
            {
                v8f acc;
                #pragma unroll
                for (int r = 0; r < 8; ++r) acc[r] = bias;
                const int arow = 4 * 16 + lr;
                #pragma unroll
                for (int kk = 0; kk < NKK; ++kk) {
                    const int kb = kk * 32 + lh * 16;
                    const __bf16* ap = Xs + arow * XPITCH + kb;
                    V16 a;
                    a.u[0] = *(const uint4*)ap;
                    a.u[1] = *(const uint4*)(ap + 8);
                    acc = __builtin_amdgcn_wmma_f32_16x16x32_bf16(
                            false, a.v, false, Bp[kk], (short)0, acc, false, false);
                }
                #pragma unroll
                for (int r = 0; r < 8; ++r) {
                    const int row = 4 * 16 + lh * 8 + r;
                    QKVs[row * QKVPITCH + col] = to_bf16(acc[r]);
                }
            }
        }
    }
    __syncthreads();

    // X region dead; reuse as f32 scratch: s[16][256] + hidden[16][16]
    float* sbuf = (float*)smem;
    float* hid  = sbuf + SPB * OUT_DIM;

    // ---------------- Phase 2: attention -> s = mean_l z (16 lanes/sample) ----
    {
        const int s    = tid >> 4;
        const int tsub = tid & 15;
        const int h    = tsub >> 2;
        const int q    = tsub & 3;
        const int d0   = q * 16;
        const __bf16* Qb = QKVs + (s * L_DIM) * QKVPITCH + h * DH_DIM + d0;
        const __bf16* Kb = Qb + OUT_DIM;
        const __bf16* Vb = Qb + 2 * OUT_DIM;

        float Qr[L_DIM][16];
        #pragma unroll
        for (int l = 0; l < L_DIM; ++l)
            #pragma unroll
            for (int i = 0; i < 16; ++i)
                Qr[l][i] = fr_bf16(Qb[l * QKVPITCH + i]);

        float att[L_DIM][L_DIM];
        #pragma unroll
        for (int m = 0; m < L_DIM; ++m) {
            float kv[16];
            #pragma unroll
            for (int i = 0; i < 16; ++i) kv[i] = fr_bf16(Kb[m * QKVPITCH + i]);
            #pragma unroll
            for (int l = 0; l < L_DIM; ++l) {
                float acc = 0.f;
                #pragma unroll
                for (int i = 0; i < 16; ++i) acc += Qr[l][i] * kv[i];
                att[l][m] = acc;
            }
        }
        #pragma unroll
        for (int l = 0; l < L_DIM; ++l)
            #pragma unroll
            for (int m = 0; m < L_DIM; ++m) {
                float v = att[l][m];
                v += __shfl_xor(v, 1, 32);
                v += __shfl_xor(v, 2, 32);
                att[l][m] = v * SCALE_F;
            }
        float w[L_DIM] = {0.f, 0.f, 0.f, 0.f, 0.f};
        #pragma unroll
        for (int l = 0; l < L_DIM; ++l) {
            float mx = att[l][0];
            #pragma unroll
            for (int m = 1; m < L_DIM; ++m) mx = fmaxf(mx, att[l][m]);
            float e[L_DIM]; float sum = 0.f;
            #pragma unroll
            for (int m = 0; m < L_DIM; ++m) { e[m] = __expf(att[l][m] - mx); sum += e[m]; }
            const float inv = 1.f / sum;
            #pragma unroll
            for (int m = 0; m < L_DIM; ++m) w[m] += e[m] * inv;
        }
        #pragma unroll
        for (int m = 0; m < L_DIM; ++m) w[m] *= (1.f / (float)L_DIM);

        float sb[16];
        #pragma unroll
        for (int i = 0; i < 16; ++i) sb[i] = 0.f;
        #pragma unroll
        for (int m = 0; m < L_DIM; ++m)
            #pragma unroll
            for (int i = 0; i < 16; ++i)
                sb[i] += w[m] * fr_bf16(Vb[m * QKVPITCH + i]);

        #pragma unroll
        for (int i = 0; i < 16; ++i)
            sbuf[s * OUT_DIM + h * DH_DIM + d0 + i] = sb[i];
    }
    __syncthreads();

    // ---------------- Phase 3a: SE hidden = relu(s @ W1 + b1) -----------------
    {
        const int s = tid >> 4;
        const int r = tid & 15;
        float a = b1[r];
        const float* sv = sbuf + s * OUT_DIM;
        #pragma unroll 8
        for (int c = 0; c < OUT_DIM; ++c) a += sv[c] * W1[c * RED_DIM + r];
        hid[s * RED_DIM + r] = fmaxf(a, 0.f);
    }
    __syncthreads();

    // ---------------- Phase 3b: out = sigmoid(hidden @ W2 + b2) * s -----------
    {
        const int s = tid >> 4;
        const int r = tid & 15;
        if (block0 + s < N) {
            float hreg[RED_DIM];
            #pragma unroll
            for (int i = 0; i < RED_DIM; ++i) hreg[i] = hid[s * RED_DIM + i];
            float* orow = out + (size_t)(block0 + s) * OUT_DIM;
            #pragma unroll
            for (int j = 0; j < 16; ++j) {
                const int c = r * 16 + j;
                float o = b2[c];
                #pragma unroll
                for (int rr = 0; rr < RED_DIM; ++rr) o += hreg[rr] * W2[rr * OUT_DIM + c];
                const float g = 1.f / (1.f + __expf(-o));
                orow[c] = g * sbuf[s * OUT_DIM + c];
            }
        }
    }
}

extern "C" void kernel_launch(void* const* d_in, const int* in_sizes, int n_in,
                              void* d_out, int out_size, void* d_ws, size_t ws_size,
                              hipStream_t stream) {
    const float* x  = (const float*)d_in[0];
    const float* Wq = (const float*)d_in[1];
    const float* bq = (const float*)d_in[2];
    const float* Wk = (const float*)d_in[3];
    const float* bk = (const float*)d_in[4];
    const float* Wv = (const float*)d_in[5];
    const float* bv = (const float*)d_in[6];
    const float* W1 = (const float*)d_in[7];
    const float* b1 = (const float*)d_in[8];
    const float* W2 = (const float*)d_in[9];
    const float* b2 = (const float*)d_in[10];
    float* out   = (float*)d_out;
    __bf16* wsw  = (__bf16*)d_ws;            // 393216 B of scratch

    const int N = in_sizes[0] / (L_DIM * IN_DIM);   // 100000
    const int blocks = (N + SPB - 1) / SPB;          // 6250

    wswz_kernel<<<(WSWZ_ELEMS + 255) / 256, 256, 0, stream>>>(Wq, Wk, Wv, wsw);

    hipFuncSetAttribute((const void*)mhla_fused_kernel,
                        hipFuncAttributeMaxDynamicSharedMemorySize, LDS_TOTAL);
    mhla_fused_kernel<<<blocks, 256, LDS_TOTAL, stream>>>(
        x, wsw, bq, bk, bv, W1, b1, W2, b2, out, N);
}
